// ParallelDynamicDualEncoder_62749472195483
// MI455X (gfx1250) — compile-verified, matrix operand order checked by reading, not run on hardware
//
#include <hip/hip_runtime.h>
#include <hip/hip_bf16.h>

typedef _Float16 v16h __attribute__((ext_vector_type(16)));
typedef float    v8f  __attribute__((ext_vector_type(8)));
typedef _Float16 h2   __attribute__((ext_vector_type(2)));

#define WMMA_F16(a, b, c) \
  __builtin_amdgcn_wmma_f32_16x16x32_f16(false, (a), false, (b), (short)0, (c), false, false)

// ---- problem constants -------------------------------------------------
#define TDIM   1249      // frames after 2x conv subsample
#define T1     2499      // frames after first conv
#define F1     39        // freq bins after first conv
#define F2     19        // freq bins after second conv
#define DMODEL 256
#define FFD    2048
#define K2CONV 2304      // 256*9 im2col K for conv2
#define KEMB   4864      // 256*19 embed K
#define ROWS2  (2*TDIM)  // both batches stacked

// fragment K-base for 16-bit WMMA operands (ISA 7.12.2)
__device__ __forceinline__ int frag_kbase(int v, int hi) {
  return ((v < 4) ? 0 : 16) + hi * 8 + (v & 3) * 2;
}

// ======================= small scalar kernels ===========================

__global__ void conv1_relu_f16(const float* __restrict__ xs, const float* __restrict__ w,
                               const float* __restrict__ bias, _Float16* __restrict__ out) {
  int idx = blockIdx.x * 256 + threadIdx.x;
  const int total = 256 * T1 * F1;
  if (idx >= total) return;
  int c = idx / (T1 * F1);
  int r = idx - c * (T1 * F1);
  int t = r / F1, f = r - t * F1;
  float acc = bias[c];
#pragma unroll
  for (int kh = 0; kh < 3; ++kh)
#pragma unroll
    for (int kw = 0; kw < 3; ++kw)
      acc += xs[(2 * t + kh) * 80 + (2 * f + kw)] * w[c * 9 + kh * 3 + kw];
  out[idx] = (_Float16)fmaxf(acc, 0.f);
}

// Convert an f32 weight matrix into f16 *pre-swizzled to the WMMA B-fragment
// layout*: dst[((kblk*(N/16)+ntile)*32 + lane)*16 + e] = B[k][n] with
// v=e/2, hi=lane/16, k = kblk*32 + frag_kbase(v,hi) + (e&1), n = ntile*16+lane%16.
// transposed==0: src is (K,N) row-major; transposed==1: src is (N,K).
__global__ void cvt_swizzleB(const float* __restrict__ src, _Float16* __restrict__ dst,
                             int K, int N, int transposed) {
  int i = blockIdx.x * 256 + threadIdx.x;
  if (i >= K * N) return;
  int e = i & 15;
  int lane = (i >> 4) & 31;
  int rest = i >> 9;
  int ntiles = N >> 4;
  int ntile = rest % ntiles;
  int kblk = rest / ntiles;
  int v = e >> 1, hi = lane >> 4, ln = lane & 15;
  int k = kblk * 32 + frag_kbase(v, hi) + (e & 1);
  int n = ntile * 16 + ln;
  float val = transposed ? src[(size_t)n * K + k] : src[(size_t)k * N + n];
  dst[i] = (_Float16)val;
}

// embed input: embin[t][c*19+f] = conv2out[(t*19+f)*256 + c]
__global__ void build_embed_input(const float* __restrict__ G, _Float16* __restrict__ embin) {
  int idx = blockIdx.x * 256 + threadIdx.x;
  if (idx >= TDIM * KEMB) return;
  int t = idx / KEMB, j = idx - t * KEMB;
  int c = j / F2, f = j - c * F2;
  embin[idx] = (_Float16)G[(size_t)(t * F2 + f) * DMODEL + c];
}

// x2[b][t][d] = E[t][d]*sqrt(256) + posenc(t,d), duplicated to both batches
__global__ void posenc_dup(const float* __restrict__ E, float* __restrict__ X2) {
  int idx = blockIdx.x * 256 + threadIdx.x;
  if (idx >= TDIM * DMODEL) return;
  int t = idx >> 8, d = idx & 255;
  int i2 = d & ~1;
  float freq = __expf(-(float)i2 * (9.210340371976184f / 256.f));  // ln(10000)/256
  float ang = (float)t * freq;
  float pe = (d & 1) ? __cosf(ang) : __sinf(ang);
  float y = E[idx] * 16.f + pe;
  X2[idx] = y;
  X2[TDIM * DMODEL + idx] = y;
}

__global__ void layernorm(const float* __restrict__ X, const float* __restrict__ g,
                          const float* __restrict__ bb, float* __restrict__ Yf,
                          _Float16* __restrict__ Yh) {
  __shared__ float sb[256];
  int row = blockIdx.x, t = threadIdx.x;
  float v = X[(size_t)row * 256 + t];
  sb[t] = v;
  __syncthreads();
  for (int s = 128; s > 0; s >>= 1) { if (t < s) sb[t] += sb[t + s]; __syncthreads(); }
  float mu = sb[0] * (1.f / 256.f);
  __syncthreads();
  float d = v - mu;
  sb[t] = d * d;
  __syncthreads();
  for (int s = 128; s > 0; s >>= 1) { if (t < s) sb[t] += sb[t + s]; __syncthreads(); }
  float var = sb[0] * (1.f / 256.f);
  float y = d * rsqrtf(var + 1e-12f) * g[t] + bb[t];
  if (Yf) Yf[(size_t)row * 256 + t] = y;
  if (Yh) Yh[(size_t)row * 256 + t] = (_Float16)y;
}

__global__ void write_msub(const unsigned char* __restrict__ masks, float* __restrict__ out) {
  int j = blockIdx.x * 256 + threadIdx.x;
  if (j < TDIM) out[j] = masks[4 * j] ? 1.f : 0.f;
}

// ======================= generic WMMA GEMM ==============================
// C[M,N] = A[M,K](f16) * Bsw[K,N](f16, fragment-swizzled) + bias[N]
// (+residual) (relu?), f32 out, optional f16 mirror.
// Block tile: 64(M) x 128(N); 8 waves; wave = 1 A fragment x 4 B fragments.
// AMODE 1 = implicit im2col of conv1 output (A = x1h, layout [ci][T1][F1],
// K = ci*9+kh*3+kw, m = to*19+fo).
template <int AMODE>
__global__ void wmma_gemm(const _Float16* __restrict__ A, const _Float16* __restrict__ Bsw,
                          const float* __restrict__ bias, const float* __restrict__ residual,
                          float* __restrict__ C, _Float16* __restrict__ C16,
                          int M, int N, int K, int relu) {
  __shared__ _Float16 As[64][40];  // 64 x 32 (+pad, 80B rows keep 16B chunks aligned)
  int tid = threadIdx.x;
  int lane = tid & 31, wid = tid >> 5;
  int hi = lane >> 4, ln = lane & 15;
  int bm = blockIdx.y * 64, bn = blockIdx.x * 128;
  int tm = wid & 3, tnb = (wid >> 2) * 4;
  v8f acc[4] = {};
  int arow = tid >> 2, acol = (tid & 3) * 8;
  int ntiles = N >> 4;
  // per-wave B fragment stream base (four adjacent n-tiles)
  size_t btile = (size_t)((bn >> 4) + tnb) * 512 + (lane << 4);

  for (int k0 = 0; k0 < K; k0 += 32) {
    // ---- stage A tile (64x32) in LDS, vectorized where possible --------
    int gm = bm + arow;
    if (AMODE == 0) {
      uint4 val = make_uint4(0u, 0u, 0u, 0u);
      if (gm < M) val = *(const uint4*)(A + (size_t)gm * K + k0 + acol);
      *(uint4*)&As[arow][acol] = val;
    } else {
#pragma unroll
      for (int e = 0; e < 8; ++e) {
        _Float16 val = (_Float16)0.f;
        int k = k0 + acol + e;
        if (gm < M) {
          int to = gm / F2, fo = gm - to * F2;
          int ci = k / 9, rr = k - ci * 9, kh = rr / 3, kw = rr - kh * 3;
          val = A[(size_t)(ci * T1 + 2 * to + kh) * F1 + (2 * fo + kw)];
        }
        As[arow][acol + e] = val;
      }
    }
    __syncthreads();

    // ---- B fragments: contiguous 32B chunks per lane per tile ----------
    const _Float16* bp = Bsw + (size_t)(k0 >> 5) * ntiles * 512 + btile;
    if (k0 + 32 < K)  // pull next k-block toward the WGP (global_prefetch_b8)
      __builtin_prefetch(bp + (size_t)ntiles * 512, 0, 1);
    v16h bf0 = *(const v16h*)bp;
    v16h bf1 = *(const v16h*)(bp + 512);
    v16h bf2 = *(const v16h*)(bp + 1024);
    v16h bf3 = *(const v16h*)(bp + 1536);

    // ---- A fragment from LDS (pairs contiguous along K) ----------------
    v16h af;
#pragma unroll
    for (int v = 0; v < 8; ++v) {
      int kb = frag_kbase(v, hi);
      h2 p = *(const h2*)(&As[tm * 16 + ln][kb]);
      af[2 * v] = p[0]; af[2 * v + 1] = p[1];
    }
    acc[0] = WMMA_F16(af, bf0, acc[0]);
    acc[1] = WMMA_F16(af, bf1, acc[1]);
    acc[2] = WMMA_F16(af, bf2, acc[2]);
    acc[3] = WMMA_F16(af, bf3, acc[3]);
    __syncthreads();
  }

#pragma unroll
  for (int t = 0; t < 4; ++t) {
    int gn = bn + (tnb + t) * 16 + ln;
#pragma unroll
    for (int r = 0; r < 8; ++r) {
      int gm = bm + tm * 16 + r + 8 * hi;
      if (gm < M) {
        float val = acc[t][r] + bias[gn];
        if (relu) val = fmaxf(val, 0.f);
        if (residual) val += residual[(size_t)gm * N + gn];
        if (C)   C[(size_t)gm * N + gn] = val;
        if (C16) C16[(size_t)gm * N + gn] = (_Float16)val;
      }
    }
  }
}

// ======================= flash attention (WMMA) =========================
// Block = 64 threads = 2 waves per (32-query block, head, batch). The two
// waves share LDS-staged K/V tiles; each wave owns one 16-query tile.
// dk = 64. Streaming softmax with fixed max = 0 (scores here are O(1);
// masked scores are -1e9 and underflow exp to exactly 0, matching the
// reference's zeroing). Row sums accumulate via a WMMA against a
// ones-column B matrix, so the inner loop has no cross-lane reductions.
__global__ void flash_attn_wmma(const _Float16* __restrict__ Qh, const _Float16* __restrict__ Kh,
                                const _Float16* __restrict__ Vh,
                                const unsigned char* __restrict__ masks,
                                _Float16* __restrict__ Oh, int T) {
  __shared__ _Float16 Klds[32][72];  // 32 keys x 64 dk (+pad)
  __shared__ _Float16 Vlds[32][72];  // 32 keys x 64 dk (+pad)
  __shared__ _Float16 Ps[32][40];    // per-wave 16 q-rows x 32 keys (+pad)
  int tid = threadIdx.x;
  int lane = tid & 31;
  int qt = tid >> 5;                 // which 16-query tile this wave owns
  int hi = lane >> 4, ln = lane & 15;
  int qb = blockIdx.x, h = blockIdx.y, b = blockIdx.z;
  size_t base = (size_t)b * T * DMODEL + h * 64;
  int frow = tid >> 3;               // 0..7  (8 rows per pass, 64 threads)
  int fcol = (tid & 7) * 8;          // 0..56 (8 f16 = 16B chunks)

  // Q fragments (A-layout): m = ln, dk split 0-31 / 32-63
  int qr = qb * 32 + qt * 16 + ln; if (qr > T - 1) qr = T - 1;
  const _Float16* qp = Qh + base + (size_t)qr * DMODEL;
  v16h q0, q1;
#pragma unroll
  for (int v = 0; v < 8; ++v) {
    int kb = frag_kbase(v, hi);
    h2 p0 = *(const h2*)(qp + kb);
    h2 p1 = *(const h2*)(qp + 32 + kb);
    q0[2 * v] = p0[0]; q0[2 * v + 1] = p0[1];
    q1[2 * v] = p1[0]; q1[2 * v + 1] = p1[1];
  }

  // ones-column B fragment: B[k][0] = 1, other columns 0
  v16h vones;
  {
    _Float16 one = (_Float16)((ln == 0) ? 1.f : 0.f);
#pragma unroll
    for (int e = 0; e < 16; ++e) vones[e] = one;
  }

  v8f o[4] = {};
  v8f osum = {};  // column 0 accumulates per-row sum of P

  for (int kb0 = 0; kb0 < T; kb0 += 32) {
    // ---- stage K/V tiles (coalesced b128, shared by both waves) --------
#pragma unroll
    for (int i = 0; i < 4; ++i) {
      int row = i * 8 + frow;
      int key = kb0 + row; if (key > T - 1) key = T - 1;
      *(uint4*)&Klds[row][fcol] = *(const uint4*)(Kh + base + (size_t)key * DMODEL + fcol);
      *(uint4*)&Vlds[row][fcol] = *(const uint4*)(Vh + base + (size_t)key * DMODEL + fcol);
    }
    __syncthreads();

    // ---- S = Q K^T: build K^T fragments tile-by-tile to cap registers --
    v8f s0 = {}, s1 = {};
    {
      v16h kt0, kt1;  // key-tile 0: dk halves
#pragma unroll
      for (int v = 0; v < 8; ++v) {
        int kb = frag_kbase(v, hi);
        h2 a = *(const h2*)(&Klds[ln][kb]);
        h2 c = *(const h2*)(&Klds[ln][32 + kb]);
        kt0[2 * v] = a[0]; kt0[2 * v + 1] = a[1];
        kt1[2 * v] = c[0]; kt1[2 * v + 1] = c[1];
      }
      s0 = WMMA_F16(q0, kt0, s0);
      s0 = WMMA_F16(q1, kt1, s0);
    }
    {
      v16h kt0, kt1;  // key-tile 1: dk halves
#pragma unroll
      for (int v = 0; v < 8; ++v) {
        int kb = frag_kbase(v, hi);
        h2 d = *(const h2*)(&Klds[16 + ln][kb]);
        h2 e = *(const h2*)(&Klds[16 + ln][32 + kb]);
        kt0[2 * v] = d[0]; kt0[2 * v + 1] = d[1];
        kt1[2 * v] = e[0]; kt1[2 * v + 1] = e[1];
      }
      s1 = WMMA_F16(q0, kt0, s1);
      s1 = WMMA_F16(q1, kt1, s1);
    }

    // ---- exp + mask, store P (no cross-lane ops) -----------------------
    int key0 = kb0 + ln, key1 = kb0 + 16 + ln;
    bool mk0 = (key0 < T) && (masks[4 * key0] != 0);
    bool mk1 = (key1 < T) && (masks[4 * key1] != 0);
#pragma unroll
    for (int r = 0; r < 8; ++r) {
      int m = r + 8 * hi;
      int qg = qb * 32 + qt * 16 + m;
      float sc0 = (float)s0[r] * 0.125f;
      float sc1 = (float)s1[r] * 0.125f;
      bool ok0 = mk0 && (qg < T) && (b == 0 || (key0 >> 4) <= (qg >> 4));
      bool ok1 = mk1 && (qg < T) && (b == 0 || (key1 >> 4) <= (qg >> 4));
      if (!ok0) sc0 = -1.0e9f;
      if (!ok1) sc1 = -1.0e9f;
      Ps[qt * 16 + m][ln]      = (_Float16)__expf(sc0);
      Ps[qt * 16 + m][16 + ln] = (_Float16)__expf(sc1);
    }

    // ---- P fragment (A-layout over 32 keys; own rows only) -------------
    v16h pf;
#pragma unroll
    for (int v = 0; v < 8; ++v) {
      int kb = frag_kbase(v, hi);
      h2 p = *(const h2*)(&Ps[qt * 16 + ln][kb]);
      pf[2 * v] = p[0]; pf[2 * v + 1] = p[1];
    }
    // ---- O += P*V and rowsum += P*ones ---------------------------------
#pragma unroll
    for (int j = 0; j < 4; ++j) {
      v16h vf;
#pragma unroll
      for (int v = 0; v < 8; ++v) {
        int kb = frag_kbase(v, hi);
        vf[2 * v]     = Vlds[kb][j * 16 + ln];
        vf[2 * v + 1] = Vlds[kb + 1][j * 16 + ln];
      }
      o[j] = WMMA_F16(pf, vf, o[j]);
    }
    osum = WMMA_F16(pf, vones, osum);
    __syncthreads();
  }

  // ---- epilogue: broadcast row sums from column-0 lanes, normalize -----
#pragma unroll
  for (int r = 0; r < 8; ++r) {
    int m = r + 8 * hi;
    int qg = qb * 32 + qt * 16 + m;
    float rs = __shfl(osum[r], hi * 16, 32);  // lane holding column 0 of this row
    float inv = 1.0f / fmaxf(rs, 1e-30f);
    if (qg < T) {
#pragma unroll
      for (int j = 0; j < 4; ++j)
        Oh[base + (size_t)qg * DMODEL + j * 16 + ln] = (_Float16)((float)o[j][r] * inv);
    }
  }
}

// =========================== host driver ================================

extern "C" void kernel_launch(void* const* d_in, const int* in_sizes, int n_in,
                              void* d_out, int out_size, void* d_ws, size_t ws_size,
                              hipStream_t stream) {
  (void)in_sizes; (void)n_in; (void)out_size; (void)ws_size;
  const float* xs          = (const float*)d_in[0];
  const unsigned char* msk = (const unsigned char*)d_in[1];
  const float* c1w = (const float*)d_in[2];
  const float* c1b = (const float*)d_in[3];
  const float* c2w = (const float*)d_in[4];
  const float* c2b = (const float*)d_in[5];
  const float* ew  = (const float*)d_in[6];
  const float* eb  = (const float*)d_in[7];
  const float* fg  = (const float*)d_in[8];
  const float* fb  = (const float*)d_in[9];

  const int M2 = TDIM * F2;  // 23731 conv2 GEMM rows

  size_t off = 0;
  auto carve = [&](size_t bytes) -> void* {
    void* p = (char*)d_ws + off;
    off += (bytes + 255) & ~(size_t)255;
    return p;
  };
  _Float16* x1h    = (_Float16*)carve((size_t)256 * T1 * F1 * 2);
  _Float16* c2wh   = (_Float16*)carve((size_t)K2CONV * DMODEL * 2);
  float*    g2f    = (float*)carve((size_t)M2 * DMODEL * 4);
  _Float16* embinh = (_Float16*)carve((size_t)TDIM * KEMB * 2);
  _Float16* ewh    = (_Float16*)carve((size_t)KEMB * DMODEL * 2);
  float*    Ef     = (float*)carve((size_t)TDIM * DMODEL * 4);
  float*    x2     = (float*)carve((size_t)ROWS2 * DMODEL * 4);
  _Float16* xnh    = (_Float16*)carve((size_t)ROWS2 * DMODEL * 2);
  _Float16* qh     = (_Float16*)carve((size_t)ROWS2 * DMODEL * 2);
  _Float16* kh     = (_Float16*)carve((size_t)ROWS2 * DMODEL * 2);
  _Float16* vh     = (_Float16*)carve((size_t)ROWS2 * DMODEL * 2);
  _Float16* aoh    = (_Float16*)carve((size_t)ROWS2 * DMODEL * 2);
  _Float16* ffh    = (_Float16*)carve((size_t)ROWS2 * FFD * 2);
  _Float16* wqh    = (_Float16*)carve((size_t)DMODEL * DMODEL * 2);
  _Float16* wkh    = (_Float16*)carve((size_t)DMODEL * DMODEL * 2);
  _Float16* wvh    = (_Float16*)carve((size_t)DMODEL * DMODEL * 2);
  _Float16* woh    = (_Float16*)carve((size_t)DMODEL * DMODEL * 2);
  _Float16* w1h    = (_Float16*)carve((size_t)DMODEL * FFD * 2);
  _Float16* w2h    = (_Float16*)carve((size_t)FFD * DMODEL * 2);

  // ---- front end -------------------------------------------------------
  { int n = 256 * T1 * F1;
    conv1_relu_f16<<<(n + 255) / 256, 256, 0, stream>>>(xs, c1w, c1b, x1h); }
  { int n = K2CONV * DMODEL;  // conv2 weight (256,2304) -> swizzled (2304,256)
    cvt_swizzleB<<<(n + 255) / 256, 256, 0, stream>>>(c2w, c2wh, K2CONV, DMODEL, 1); }
  { dim3 g(DMODEL / 128, (M2 + 63) / 64);  // conv2 as implicit-im2col WMMA GEMM
    wmma_gemm<1><<<g, 256, 0, stream>>>(x1h, c2wh, c2b, nullptr, g2f, nullptr,
                                        M2, DMODEL, K2CONV, 1); }
  { int n = TDIM * KEMB;
    build_embed_input<<<(n + 255) / 256, 256, 0, stream>>>(g2f, embinh); }
  { int n = KEMB * DMODEL;
    cvt_swizzleB<<<(n + 255) / 256, 256, 0, stream>>>(ew, ewh, KEMB, DMODEL, 0); }
  { dim3 g(DMODEL / 128, (TDIM + 63) / 64);
    wmma_gemm<0><<<g, 256, 0, stream>>>(embinh, ewh, eb, nullptr, Ef, nullptr,
                                        TDIM, DMODEL, KEMB, 0); }
  { int n = TDIM * DMODEL;
    posenc_dup<<<(n + 255) / 256, 256, 0, stream>>>(Ef, x2); }

  // ---- transformer layers ---------------------------------------------
  dim3 gp(DMODEL / 128, (ROWS2 + 63) / 64);
  dim3 gf(FFD / 128, (ROWS2 + 63) / 64);
  dim3 ga((TDIM + 31) / 32, 4, 2);
  for (int l = 0; l < 6; ++l) {
    const float* g1  = (const float*)d_in[10 + l * 16 + 0];
    const float* b1  = (const float*)d_in[10 + l * 16 + 1];
    const float* g2  = (const float*)d_in[10 + l * 16 + 2];
    const float* b2  = (const float*)d_in[10 + l * 16 + 3];
    const float* wq  = (const float*)d_in[10 + l * 16 + 4];
    const float* bq  = (const float*)d_in[10 + l * 16 + 5];
    const float* wk  = (const float*)d_in[10 + l * 16 + 6];
    const float* bk  = (const float*)d_in[10 + l * 16 + 7];
    const float* wv  = (const float*)d_in[10 + l * 16 + 8];
    const float* bv  = (const float*)d_in[10 + l * 16 + 9];
    const float* wo  = (const float*)d_in[10 + l * 16 + 10];
    const float* bo  = (const float*)d_in[10 + l * 16 + 11];
    const float* w1  = (const float*)d_in[10 + l * 16 + 12];
    const float* b1f = (const float*)d_in[10 + l * 16 + 13];
    const float* w2  = (const float*)d_in[10 + l * 16 + 14];
    const float* b2f = (const float*)d_in[10 + l * 16 + 15];

    { int n = DMODEL * DMODEL;
      cvt_swizzleB<<<(n + 255) / 256, 256, 0, stream>>>(wq, wqh, DMODEL, DMODEL, 0);
      cvt_swizzleB<<<(n + 255) / 256, 256, 0, stream>>>(wk, wkh, DMODEL, DMODEL, 0);
      cvt_swizzleB<<<(n + 255) / 256, 256, 0, stream>>>(wv, wvh, DMODEL, DMODEL, 0);
      cvt_swizzleB<<<(n + 255) / 256, 256, 0, stream>>>(wo, woh, DMODEL, DMODEL, 0); }
    { int n = DMODEL * FFD;
      cvt_swizzleB<<<(n + 255) / 256, 256, 0, stream>>>(w1, w1h, DMODEL, FFD, 0);
      cvt_swizzleB<<<(n + 255) / 256, 256, 0, stream>>>(w2, w2h, FFD, DMODEL, 0); }

    layernorm<<<ROWS2, 256, 0, stream>>>(x2, g1, b1, nullptr, xnh);
    wmma_gemm<0><<<gp, 256, 0, stream>>>(xnh, wqh, bq, nullptr, nullptr, qh,
                                         ROWS2, DMODEL, DMODEL, 0);
    wmma_gemm<0><<<gp, 256, 0, stream>>>(xnh, wkh, bk, nullptr, nullptr, kh,
                                         ROWS2, DMODEL, DMODEL, 0);
    wmma_gemm<0><<<gp, 256, 0, stream>>>(xnh, wvh, bv, nullptr, nullptr, vh,
                                         ROWS2, DMODEL, DMODEL, 0);
    flash_attn_wmma<<<ga, 64, 0, stream>>>(qh, kh, vh, msk, aoh, TDIM);
    wmma_gemm<0><<<gp, 256, 0, stream>>>(aoh, woh, bo, x2, x2, nullptr,
                                         ROWS2, DMODEL, DMODEL, 0);
    layernorm<<<ROWS2, 256, 0, stream>>>(x2, g2, b2, nullptr, xnh);
    wmma_gemm<0><<<gf, 256, 0, stream>>>(xnh, w1h, b1f, nullptr, nullptr, ffh,
                                         ROWS2, FFD, DMODEL, 1);
    wmma_gemm<0><<<gp, 256, 0, stream>>>(ffh, w2h, b2f, x2, x2, nullptr,
                                         ROWS2, DMODEL, FFD, 0);
  }

  // ---- final LN + outputs ---------------------------------------------
  layernorm<<<ROWS2, 256, 0, stream>>>(x2, fg, fb, (float*)d_out, nullptr);
  write_msub<<<(TDIM + 255) / 256, 256, 0, stream>>>(msk, (float*)d_out + (size_t)ROWS2 * DMODEL);
}
